// InvariantPointAttention_36172214566966
// MI455X (gfx1250) — matrix operand order repaired and checked by reading, not compiled, not run on hardware
//
#include <hip/hip_runtime.h>
#include <math.h>

// Invariant Point Attention for MI455X (gfx1250), wave32 + bf16 WMMA.
// B=1, N=768, C1=384, C2=128, H=12, SQK=SV=16, PQK=4, PV=8.

#define HH   12
#define SQKD 16
#define SVD  16
#define PQKD 4
#define PVD  8
#define C1D  384
#define C2D  128
#define ND   768
#define FIN  2112               // H*(C2 + SV + 4*PV)
#define NN2  ((long)ND * ND)    // 589824

typedef __attribute__((ext_vector_type(16))) __bf16 v16bf;
typedef __attribute__((ext_vector_type(8)))  __bf16 v8bf;
typedef __attribute__((ext_vector_type(4)))  __bf16 v4bf;
typedef __attribute__((ext_vector_type(8)))  float  v8f;
typedef __attribute__((ext_vector_type(4)))  float  f32x4;

// ---------- bf16 conversion (RNE, integer ops only) ----------
__device__ __forceinline__ __bf16 f2bf(float f) {
  unsigned u = __builtin_bit_cast(unsigned, f);
  unsigned r = u + 0x7FFFu + ((u >> 16) & 1u);
  unsigned short h = (unsigned short)(r >> 16);
  return __builtin_bit_cast(__bf16, h);
}

__device__ __forceinline__ v16bf cat8(v8bf lo, v8bf hi) {
  return __builtin_shufflevector(lo, hi, 0, 1, 2, 3, 4, 5, 6, 7, 8, 9, 10, 11, 12, 13, 14, 15);
}

__device__ __forceinline__ v8f wmma_bf16(v16bf a, v16bf b, v8f c) {
  return __builtin_amdgcn_wmma_f32_16x16x32_bf16(false, a, false, b, (short)0, c, false, false);
}

// ---------- A fragment from bf16 row-major: two b128 loads, no conversion ----------
// ISA 7.12.2: lane m=lane&15; lanes>=16 add +8 to K; K runs [khi..khi+7], [16+khi..23+khi].
__device__ __forceinline__ v16bf load_a_bf(const __bf16* __restrict__ A, long lda,
                                           int m0, int k0, int lane) {
  const __bf16* p = A + (long)(m0 + (lane & 15)) * lda + k0 + ((lane >> 4) * 8);
  return cat8(*(const v8bf*)(p), *(const v8bf*)(p + 16));
}

// ---------- B fragment from bf16 B^T (row n holds K contiguous): two b128 loads ----------
__device__ __forceinline__ v16bf load_bT_bf(const __bf16* __restrict__ BT, long ldn,
                                            int n0, int k0, int lane) {
  const __bf16* p = BT + (long)(n0 + (lane & 15)) * ldn + k0 + ((lane >> 4) * 8);
  return cat8(*(const v8bf*)(p), *(const v8bf*)(p + 16));
}

// ---------- A fragment from fp32 row-major (act_2d stream), with conversion ----------
__device__ __forceinline__ v16bf load_a_f32(const float* __restrict__ A, long lda,
                                            long m0, int k0, int lane) {
  const float* p = A + (m0 + (lane & 15)) * lda + k0 + ((lane >> 4) * 8);
  f32x4 x0 = *(const f32x4*)(p);
  f32x4 x1 = *(const f32x4*)(p + 4);
  f32x4 x2 = *(const f32x4*)(p + 16);
  f32x4 x3 = *(const f32x4*)(p + 20);
  v16bf a;
#pragma unroll
  for (int e = 0; e < 4; ++e) {
    a[e]      = f2bf(x0[e]);
    a[4 + e]  = f2bf(x1[e]);
    a[8 + e]  = f2bf(x2[e]);
    a[12 + e] = f2bf(x3[e]);
  }
  return a;
}

// ---------- B fragment, fp32 row-major B[k*ldb+n], strided scalar (unguarded) ----------
__device__ __forceinline__ v16bf load_b_strided_f32(const float* __restrict__ B, long ldb,
                                                    int n0, int k0, int lane) {
  int n  = n0 + (lane & 15);
  int kh = (lane >> 4) * 8;
  v16bf b;
#pragma unroll
  for (int e = 0; e < 16; ++e)
    b[e] = f2bf(B[(long)(k0 + kh + e + (e >= 8 ? 8 : 0)) * ldb + n]);
  return b;
}

// ---------- fp32 -> bf16 stream convert (4 elems / thread) ----------
__global__ void conv_bf16(const float* __restrict__ src, __bf16* __restrict__ dst, int n4) {
  int i = blockIdx.x * blockDim.x + threadIdx.x;
  if (i >= n4) return;
  f32x4 x = *(const f32x4*)(src + 4 * (long)i);
  v4bf y;
#pragma unroll
  for (int e = 0; e < 4; ++e) y[e] = f2bf(x[e]);
  *(v4bf*)(dst + 4 * (long)i) = y;
}

// ---------- weight transpose + zero-pad -> bf16: BT[Npad][K] from B[K][N] ----------
__global__ void transpose_pad(const float* __restrict__ B, __bf16* __restrict__ BT,
                              int K, int N, int Npad) {
  int idx = blockIdx.x * blockDim.x + threadIdx.x;
  if (idx >= Npad * K) return;
  int n = idx / K, k = idx % K;
  BT[(long)n * K + k] = (n < N) ? f2bf(B[(long)k * N + n]) : f2bf(0.0f);
}

// ---------- generic GEMM: C[M,N] = A[M,K] * BT[N,K]^T + bias (bf16 in, f32 out) ----------
__global__ void wmma_gemmT(const __bf16* __restrict__ A, const __bf16* __restrict__ BT,
                           const float* __restrict__ bias, float* __restrict__ C,
                           int M, int N, int K) {
  int ntn  = (N + 15) >> 4;
  int tm   = blockIdx.x / ntn;
  int tn   = blockIdx.x % ntn;
  int lane = threadIdx.x;
  v8f acc = {};
  for (int k0 = 0; k0 < K; k0 += 32) {
    v16bf a = load_a_bf(A, K, tm * 16, k0, lane);
    v16bf b = load_bT_bf(BT, K, tn * 16, k0, lane);
    acc = wmma_bf16(a, b, acc);
  }
  int n  = tn * 16 + (lane & 15);
  int mo = (lane >> 4) * 8;
  if (n < N) {
    float bv = bias ? bias[n] : 0.0f;
#pragma unroll
    for (int r = 0; r < 8; ++r)
      C[(long)(tm * 16 + mo + r) * N + n] = acc[r] + bv;
  }
}

// ---------- act_2d @ w_2d + b_2d -> P[h][q*N+k] fp32 ----------
__global__ void ipa_bias2d(const float* __restrict__ act2d, const __bf16* __restrict__ w2dT,
                           const float* __restrict__ b2d, float* __restrict__ P) {
  long tm  = blockIdx.x;          // 16-row tile of the 589824 (q,k) rows
  int lane = threadIdx.x;
  v8f acc = {};
  for (int k0 = 0; k0 < C2D; k0 += 32) {
    v16bf a = load_a_f32(act2d, C2D, tm * 16, k0, lane);
    v16bf b = load_bT_bf(w2dT, C2D, 0, k0, lane);
    acc = wmma_bf16(a, b, acc);
  }
  int h  = lane & 15;
  int mo = (lane >> 4) * 8;
  if (h < HH) {
    float bv = b2d[h];
#pragma unroll
    for (int r = 0; r < 8; ++r)
      P[(long)h * NN2 + tm * 16 + mo + r] = acc[r] + bv;
  }
}

// ---------- rigid apply: pt[n, h*P+p, :] = rot[n]*p + trans[n] ----------
__global__ void ipa_points(const float* __restrict__ lin, const float* __restrict__ rot,
                           const float* __restrict__ trans, float* __restrict__ pt,
                           int P, int total) {
  int idx = blockIdx.x * blockDim.x + threadIdx.x;
  if (idx >= total) return;
  int p = idx % P, h = (idx / P) % HH, n = idx / (P * HH);
  long lb = (long)n * (HH * 3 * P) + (long)h * 3 * P;
  float x = lin[lb + 0 * P + p], y = lin[lb + 1 * P + p], z = lin[lb + 2 * P + p];
  const float* R = rot + (long)n * 9;
  const float* T = trans + (long)n * 3;
  long ob = ((long)n * HH * P + (long)h * P + p) * 3;
#pragma unroll
  for (int i = 0; i < 3; ++i)
    pt[ob + i] = R[i * 3 + 0] * x + R[i * 3 + 1] * y + R[i * 3 + 2] * z + T[i];
}

// ---------- pack Qe/Ke (28->32, bf16), VcatT[h][c][k] (40->48, bf16), corrections ----------
__global__ void ipa_prep(const float* __restrict__ Sq, const float* __restrict__ Sk,
                         const float* __restrict__ Sv,
                         const float* __restrict__ qpt, const float* __restrict__ kpt,
                         const float* __restrict__ vpt, const float* __restrict__ tw,
                         __bf16* __restrict__ Qe, __bf16* __restrict__ Ke,
                         __bf16* __restrict__ VcatT, float* __restrict__ cq,
                         float* __restrict__ ck) {
  int idx = blockIdx.x * blockDim.x + threadIdx.x;
  if (idx >= ND * HH) return;
  int h = idx % HH, n = idx / HH;
  float x  = tw[h];
  float sp = (x > 20.0f) ? x : log1pf(expf(x));         // softplus
  float wh = 0.23570226039551584f * sp;                 // sqrt(1/18)
  __bf16* qe = Qe + (long)idx * 32;
  __bf16* ke = Ke + (long)idx * 32;
#pragma unroll
  for (int c = 0; c < 16; ++c) {
    qe[c] = f2bf(Sq[(long)n * (HH * SQKD) + h * SQKD + c] * 0.25f);  // sqrt(1/SQK)
    ke[c] = f2bf(Sk[(long)n * (HH * SQKD) + h * SQKD + c]);
  }
  float sq = 0.0f, sk = 0.0f;
#pragma unroll
  for (int p = 0; p < PQKD; ++p)
#pragma unroll
    for (int d = 0; d < 3; ++d) {
      float qv = qpt[((long)n * HH * PQKD + h * PQKD + p) * 3 + d];
      float kv = kpt[((long)n * HH * PQKD + h * PQKD + p) * 3 + d];
      qe[16 + p * 3 + d] = f2bf(qv * wh);
      ke[16 + p * 3 + d] = f2bf(kv);
      sq += qv * qv;
      sk += kv * kv;
    }
#pragma unroll
  for (int c = 28; c < 32; ++c) { qe[c] = f2bf(0.0f); ke[c] = f2bf(0.0f); }
  cq[idx] = -0.5f * wh * sq;
  ck[idx] = -0.5f * wh * sk;
#pragma unroll
  for (int c = 0; c < 16; ++c)
    VcatT[((long)h * 48 + c) * ND + n] = f2bf(Sv[(long)n * (HH * SVD) + h * SVD + c]);
#pragma unroll
  for (int p = 0; p < PVD; ++p)
#pragma unroll
    for (int d = 0; d < 3; ++d)
      VcatT[((long)h * 48 + 16 + p * 3 + d) * ND + n] =
          f2bf(vpt[((long)n * HH * PVD + h * PVD + p) * 3 + d]);
#pragma unroll
  for (int c = 40; c < 48; ++c) VcatT[((long)h * 48 + c) * ND + n] = f2bf(0.0f);
}

// ---------- logits: P[h][q][k] += Qe_h.Ke_h + cq + ck; mask; *1/sqrt(3) ----------
__global__ void ipa_logits(const __bf16* __restrict__ Qe, const __bf16* __restrict__ Ke,
                           const float* __restrict__ cq, const float* __restrict__ ck,
                           const float* __restrict__ mask, float* __restrict__ P) {
  int tn = blockIdx.x, tm = blockIdx.y, h = blockIdx.z;
  int lane = threadIdx.x;
  v8f acc = {};
  v16bf a = load_a_bf(Qe + h * 32, (long)HH * 32, tm * 16, 0, lane);
  v16bf b = load_bT_bf(Ke + h * 32, (long)HH * 32, tn * 16, 0, lane);
  acc = wmma_bf16(a, b, acc);
  int k   = tn * 16 + (lane & 15);
  int mo  = (lane >> 4) * 8;
  float mk  = mask[k];
  float ckv = ck[k * HH + h];
  float* Ph = P + (long)h * NN2;
  const float inv_sqrt3 = 0.57735026918962576f;
#pragma unroll
  for (int r = 0; r < 8; ++r) {
    int q = tm * 16 + mo + r;
    float* pp = Ph + (long)q * ND + k;
    float v = acc[r] + cq[q * HH + h] + ckv + *pp;
    v -= 1e5f * (1.0f - mask[q] * mk);
    *pp = v * inv_sqrt3;
  }
}

// ---------- softmax over keys: single pass, row in registers; writes bf16 probs ----------
__global__ void ipa_softmax(const float* __restrict__ P, __bf16* __restrict__ Pb) {
  long row = blockIdx.x;                   // h*ND + q
  const float* base = P + row * ND;
  __bf16* ob = Pb + row * ND;
  int lane = threadIdx.x;
  float v[24];
#pragma unroll
  for (int i = 0; i < 24; ++i) v[i] = base[lane + 32 * i];
  float m = -1e30f;
#pragma unroll
  for (int i = 0; i < 24; ++i) m = fmaxf(m, v[i]);
#pragma unroll
  for (int s = 16; s > 0; s >>= 1) m = fmaxf(m, __shfl_xor(m, s, 32));
  float sum = 0.0f;
#pragma unroll
  for (int i = 0; i < 24; ++i) { v[i] = expf(v[i] - m); sum += v[i]; }
#pragma unroll
  for (int s = 16; s > 0; s >>= 1) sum += __shfl_xor(sum, s, 32);
  float inv = 1.0f / sum;
#pragma unroll
  for (int i = 0; i < 24; ++i) ob[lane + 32 * i] = f2bf(v[i] * inv);
}

// ---------- per-head attn x [v_scalar|v_pt]: O1[q][h][0..47] fp32 ----------
__global__ void ipa_attn_v(const __bf16* __restrict__ Pb, const __bf16* __restrict__ VcatT,
                           float* __restrict__ O1) {
  int tn = blockIdx.x, tm = blockIdx.y, h = blockIdx.z;
  int lane = threadIdx.x;
  const __bf16* Ah = Pb + (long)h * NN2;
  const __bf16* Bh = VcatT + (long)h * 48 * ND;
  v8f acc = {};
  for (int k0 = 0; k0 < ND; k0 += 32) {
    v16bf a = load_a_bf(Ah, ND, tm * 16, k0, lane);
    v16bf b = load_bT_bf(Bh, ND, tn * 16, k0, lane);
    acc = wmma_bf16(a, b, acc);
  }
  int c  = tn * 16 + (lane & 15);
  int mo = (lane >> 4) * 8;
#pragma unroll
  for (int r = 0; r < 8; ++r)
    O1[((long)(tm * 16 + mo + r) * HH + h) * 48 + c] = acc[r];
}

// ---------- per-q attn x act_2d -> Fb[q][576 + h*128 + c] (bf16) ----------
__global__ void ipa_attn_2d(const __bf16* __restrict__ Pb, const float* __restrict__ act2d,
                            __bf16* __restrict__ Fb) {
  int tn = blockIdx.x;                 // c tile 0..7
  int q  = blockIdx.y;
  int lane = threadIdx.x;
  int hrow = (lane & 15) < HH ? (lane & 15) : (HH - 1);  // clamp pad rows (not stored)
  const __bf16* Abase = Pb + (long)q * ND + (long)hrow * NN2;
  const float* Bq = act2d + (long)q * ND * C2D;
  v8f acc = {};
  for (int k0 = 0; k0 < ND; k0 += 32) {
    if (k0 + 64 < ND)
      __builtin_prefetch(Bq + (long)(k0 + 64) * C2D + tn * 16 + (lane & 15), 0, 1);
    const __bf16* p = Abase + k0 + ((lane >> 4) * 8);
    v16bf a = cat8(*(const v8bf*)(p), *(const v8bf*)(p + 16));
    v16bf b = load_b_strided_f32(Bq, C2D, tn * 16, k0, lane);
    acc = wmma_bf16(a, b, acc);
  }
  int c  = tn * 16 + (lane & 15);
  int mo = (lane >> 4) * 8;
#pragma unroll
  for (int r = 0; r < 8; ++r) {
    int h = mo + r;
    if (h < HH) Fb[(long)q * FIN + 576 + h * C2D + c] = f2bf(acc[r]);
  }
}

// ---------- invert-apply, norms, scalar copies into final_act (bf16) ----------
__global__ void ipa_assemble(const float* __restrict__ O1, const float* __restrict__ rot,
                             const float* __restrict__ trans, __bf16* __restrict__ Fb) {
  int idx = blockIdx.x * blockDim.x + threadIdx.x;
  if (idx >= ND * HH * PVD) return;
  int p = idx % PVD, h = (idx / PVD) % HH, n = idx / (PVD * HH);
  const float* o = O1 + ((long)n * HH + h) * 48;
  const float* T = trans + (long)n * 3;
  float gx = o[16 + p * 3 + 0] - T[0];
  float gy = o[16 + p * 3 + 1] - T[1];
  float gz = o[16 + p * 3 + 2] - T[2];
  const float* R = rot + (long)n * 9;
  float lx = R[0] * gx + R[3] * gy + R[6] * gz;   // rot^T
  float ly = R[1] * gx + R[4] * gy + R[7] * gz;
  float lz = R[2] * gx + R[5] * gy + R[8] * gz;
  float nrm = sqrtf(lx * lx + ly * ly + lz * lz);
  __bf16* Fr = Fb + (long)n * FIN;
  int pi = h * PVD + p;
  Fr[192 + pi] = f2bf(lx);
  Fr[288 + pi] = f2bf(ly);
  Fr[384 + pi] = f2bf(lz);
  Fr[480 + pi] = f2bf(nrm);
  int t0 = 2 * pi;
#pragma unroll
  for (int t = t0; t < t0 + 2; ++t) {
    int h2 = t / SVD, c2 = t % SVD;
    Fr[t] = f2bf(O1[((long)n * HH + h2) * 48 + c2]);
  }
}

extern "C" void kernel_launch(void* const* d_in, const int* in_sizes, int n_in,
                              void* d_out, int out_size, void* d_ws, size_t ws_size,
                              hipStream_t stream) {
  (void)in_sizes; (void)n_in; (void)out_size; (void)ws_size;
  const float* act   = (const float*)d_in[0];
  const float* act2d = (const float*)d_in[1];
  const float* mask  = (const float*)d_in[2];
  const float* rot   = (const float*)d_in[3];
  const float* trans = (const float*)d_in[4];
  const float* w_q   = (const float*)d_in[5];
  const float* w_k   = (const float*)d_in[6];
  const float* w_v   = (const float*)d_in[7];
  const float* w_qp  = (const float*)d_in[8];
  const float* b_qp  = (const float*)d_in[9];
  const float* w_kp  = (const float*)d_in[10];
  const float* b_kp  = (const float*)d_in[11];
  const float* w_vp  = (const float*)d_in[12];
  const float* b_vp  = (const float*)d_in[13];
  const float* w_2d  = (const float*)d_in[14];
  const float* b_2d  = (const float*)d_in[15];
  const float* w_fin = (const float*)d_in[16];
  const float* b_fin = (const float*)d_in[17];
  const float* tw    = (const float*)d_in[18];
  float* out = (float*)d_out;

  // ---- workspace carve (64B aligned chunks), ~58 MB ----
  char* wsb = (char*)d_ws;
  auto alloc = [&](size_t bytes) {
    char* p = wsb;
    wsb += (bytes + 63) & ~(size_t)63;
    return p;
  };
  float*  P      = (float*)alloc(sizeof(float) * HH * NN2);        // 28.3 MB
  __bf16* Pb     = (__bf16*)alloc(2 * (size_t)HH * NN2);           // 14.2 MB
  __bf16* Fb     = (__bf16*)alloc(2 * (size_t)ND * FIN);           // 3.2 MB
  float*  O1     = (float*)alloc(sizeof(float) * ND * HH * 48);
  float*  Sq     = (float*)alloc(sizeof(float) * ND * 192);
  float*  Sk     = (float*)alloc(sizeof(float) * ND * 192);
  float*  Sv     = (float*)alloc(sizeof(float) * ND * 192);
  float*  Lqp    = (float*)alloc(sizeof(float) * ND * 144);
  float*  Lkp    = (float*)alloc(sizeof(float) * ND * 144);
  float*  Lvp    = (float*)alloc(sizeof(float) * ND * 288);
  float*  qpt    = (float*)alloc(sizeof(float) * ND * 144);
  float*  kpt    = (float*)alloc(sizeof(float) * ND * 144);
  float*  vpt    = (float*)alloc(sizeof(float) * ND * 288);
  __bf16* Qeb    = (__bf16*)alloc(2 * (size_t)ND * HH * 32);
  __bf16* Keb    = (__bf16*)alloc(2 * (size_t)ND * HH * 32);
  __bf16* VcatTb = (__bf16*)alloc(2 * (size_t)HH * 48 * ND);
  float*  cq     = (float*)alloc(sizeof(float) * ND * HH);
  float*  ck     = (float*)alloc(sizeof(float) * ND * HH);
  __bf16* actb   = (__bf16*)alloc(2 * (size_t)ND * C1D);
  __bf16* wqT    = (__bf16*)alloc(2 * (size_t)192 * C1D);
  __bf16* wkT    = (__bf16*)alloc(2 * (size_t)192 * C1D);
  __bf16* wvT    = (__bf16*)alloc(2 * (size_t)192 * C1D);
  __bf16* wqpT   = (__bf16*)alloc(2 * (size_t)144 * C1D);
  __bf16* wkpT   = (__bf16*)alloc(2 * (size_t)144 * C1D);
  __bf16* wvpT   = (__bf16*)alloc(2 * (size_t)288 * C1D);
  __bf16* w2dT   = (__bf16*)alloc(2 * (size_t)16 * C2D);
  __bf16* wfinT  = (__bf16*)alloc(2 * (size_t)C1D * FIN);

  dim3 wv(32);
  auto tp = [&](const float* B, __bf16* BT, int K, int N, int Npad) {
    transpose_pad<<<(Npad * K + 255) / 256, 256, 0, stream>>>(B, BT, K, N, Npad);
  };
  auto tiles = [](int M, int N) { return dim3((unsigned)((M / 16) * ((N + 15) / 16))); };

  // 0. stage operands in bf16 (one-time)
  conv_bf16<<<(ND * C1D / 4 + 255) / 256, 256, 0, stream>>>(act, actb, ND * C1D / 4);
  tp(w_q,   wqT,   C1D, 192, 192);
  tp(w_k,   wkT,   C1D, 192, 192);
  tp(w_v,   wvT,   C1D, 192, 192);
  tp(w_qp,  wqpT,  C1D, 144, 144);
  tp(w_kp,  wkpT,  C1D, 144, 144);
  tp(w_vp,  wvpT,  C1D, 288, 288);
  tp(w_2d,  w2dT,  C2D, HH,  16);
  tp(w_fin, wfinT, FIN, C1D, C1D);

  // 1. projections (pure bf16 WMMA GEMMs)
  wmma_gemmT<<<tiles(ND, 192), wv, 0, stream>>>(actb, wqT,  nullptr, Sq,  ND, 192, C1D);
  wmma_gemmT<<<tiles(ND, 192), wv, 0, stream>>>(actb, wkT,  nullptr, Sk,  ND, 192, C1D);
  wmma_gemmT<<<tiles(ND, 192), wv, 0, stream>>>(actb, wvT,  nullptr, Sv,  ND, 192, C1D);
  wmma_gemmT<<<tiles(ND, 144), wv, 0, stream>>>(actb, wqpT, b_qp,    Lqp, ND, 144, C1D);
  wmma_gemmT<<<tiles(ND, 144), wv, 0, stream>>>(actb, wkpT, b_kp,    Lkp, ND, 144, C1D);
  wmma_gemmT<<<tiles(ND, 288), wv, 0, stream>>>(actb, wvpT, b_vp,    Lvp, ND, 288, C1D);

  // 2. rigid-apply points
  int npq = ND * HH * PQKD, npv = ND * HH * PVD;
  ipa_points<<<(npq + 255) / 256, 256, 0, stream>>>(Lqp, rot, trans, qpt, PQKD, npq);
  ipa_points<<<(npq + 255) / 256, 256, 0, stream>>>(Lkp, rot, trans, kpt, PQKD, npq);
  ipa_points<<<(npv + 255) / 256, 256, 0, stream>>>(Lvp, rot, trans, vpt, PVD,  npv);

  // 3. pack Qe/Ke/VcatT (bf16) + corrections
  ipa_prep<<<(ND * HH + 255) / 256, 256, 0, stream>>>(Sq, Sk, Sv, qpt, kpt, vpt, tw,
                                                      Qeb, Keb, VcatTb, cq, ck);

  // 4. 2d bias projection -> P[h][q][k]  (dominant HBM stream #1)
  ipa_bias2d<<<dim3((unsigned)(NN2 / 16)), wv, 0, stream>>>(act2d, w2dT, b_2d, P);

  // 5. logits + mask + scale
  ipa_logits<<<dim3(ND / 16, ND / 16, HH), wv, 0, stream>>>(Qeb, Keb, cq, ck, mask, P);

  // 6. softmax over keys -> bf16 probability planes
  ipa_softmax<<<HH * ND, wv, 0, stream>>>(P, Pb);

  // 7. attn x [v_scalar|v_pt]
  ipa_attn_v<<<dim3(3, ND / 16, HH), wv, 0, stream>>>(Pb, VcatTb, O1);

  // 8. attn x act_2d -> final_act 2d region (dominant HBM stream #2)
  ipa_attn_2d<<<dim3(C2D / 16, ND), wv, 0, stream>>>(Pb, act2d, Fb);

  // 9. invert-apply, norms, scalar copies
  ipa_assemble<<<(ND * HH * PVD + 255) / 256, 256, 0, stream>>>(O1, rot, trans, Fb);

  // 10. final projection
  wmma_gemmT<<<tiles(ND, C1D), wv, 0, stream>>>(Fb, wfinT, b_fin, out, ND, C1D, FIN);
}
